// EGNN_13305808683174
// MI455X (gfx1250) — matrix operand order repaired and checked by reading, not compiled, not run on hardware
//
#include <hip/hip_runtime.h>
#include <math.h>

// ---------------------------------------------------------------------------
// EGNN fused implementation for gfx1250 (MI455X), wave32 + WMMA bf16 + TDM.
// ---------------------------------------------------------------------------

typedef __attribute__((ext_vector_type(16))) __bf16 v16bf;
typedef __attribute__((ext_vector_type(8)))  float  v8f;
typedef __attribute__((ext_vector_type(4)))  unsigned int u32x4;
typedef __attribute__((ext_vector_type(8)))  int i32x8;
typedef __attribute__((ext_vector_type(4)))  int i32x4;

union Frag16 { v16bf v; unsigned u[8]; };

#if defined(__has_builtin)
#if __has_builtin(__builtin_amdgcn_tensor_load_to_lds) && __has_builtin(__builtin_amdgcn_s_wait_tensorcnt)
#define HAVE_TDM 1
#endif
#endif
#ifndef HAVE_TDM
#define HAVE_TDM 0
#endif

__device__ __forceinline__ unsigned short f2bf(float f) {
    unsigned u = __builtin_bit_cast(unsigned, f);
    u += 0x7fffu + ((u >> 16) & 1u);           // round-to-nearest-even
    return (unsigned short)(u >> 16);
}
__device__ __forceinline__ unsigned pack2(float a, float b) {
    return (unsigned)f2bf(a) | ((unsigned)f2bf(b) << 16);
}

// ---------------------------------------------------------------------------
// Preload 24576 uints (96KB) of packed weights global->LDS.
// Preferred path: Tensor Data Mover (one 2D descriptor: 128x192 dwords),
// issued by wave 0 (TDM ignores EXEC; TENSORcnt is per-wave so the issuing
// wave waits, then the block syncs). Fallback: cooperative copy loop.
// ---------------------------------------------------------------------------
__device__ __forceinline__ void load_weights_lds(unsigned* sW,
                                                 const unsigned* __restrict__ pkSrc) {
#if HAVE_TDM
    if (threadIdx.x < 32) {
        unsigned long long ga = (unsigned long long)(size_t)pkSrc;
        unsigned ldsOff = (unsigned)(size_t)(void*)sW;   // low 32 bits = LDS offset
        // D# group 0: count=1 | lds_addr | global_addr[56:0] | type=2
        u32x4 g0 = { 1u,
                     ldsOff,
                     (unsigned)ga,
                     (unsigned)((ga >> 32) & 0x01FFFFFFu) | (2u << 30) };
        // D# group 1: data_size=4B; tensor 128x192 dwords; tile 128x192; stride0=128
        i32x8 g1 = { (int)(2u << 16),          // data_size=2 (4 bytes)
                     (int)(128u << 16),        // tensor_dim0[15:0] = 128
                     (int)(192u << 16),        // tensor_dim1[15:0] = 192
                     (int)(128u << 16),        // tile_dim0 = 128
                     (int)192,                 // tile_dim1 = 192
                     (int)128,                 // tensor_dim0_stride = 128
                     (int)(24576u << 16),      // tensor_dim1_stride[15:0]
                     0 };
        i32x4 gz = { 0, 0, 0, 0 };
#if __clang_major__ >= 23
        i32x8 gz8 = { 0, 0, 0, 0, 0, 0, 0, 0 };
        __builtin_amdgcn_tensor_load_to_lds(g0, g1, gz, gz, gz8, 0);
#else
        __builtin_amdgcn_tensor_load_to_lds(g0, g1, gz, gz, 0);
#endif
        __builtin_amdgcn_s_wait_tensorcnt(0);
    }
    __syncthreads();
#else
    for (int i = threadIdx.x; i < 24576; i += 256) sW[i] = pkSrc[i];
    __syncthreads();
#endif
}

// ---------------------------------------------------------------------------
// Pack f32 weight matrices [K][128] into bf16 k-pair uints: pk[(k/2)*128+n] =
// {bf16(W[k][n]), bf16(W[k+1][n])} -- matches the WMMA B-operand VGPR layout.
// Layout in pk (uints): eW1(16384) eW2(8192) nW1(16384) nW2(8192) vW1(8192) cW1(8192)
// ---------------------------------------------------------------------------
__global__ void egnn_pack_kernel(const float* __restrict__ eW1,
                                 const float* __restrict__ eW2,
                                 const float* __restrict__ nW1,
                                 const float* __restrict__ nW2,
                                 const float* __restrict__ vW1,
                                 const float* __restrict__ cW1,
                                 unsigned* __restrict__ pk) {
    int i = blockIdx.x * blockDim.x + threadIdx.x;
    if (i >= 65536) return;
    const float* src; int base;
    if      (i < 16384) { src = eW1; base = 0;     }
    else if (i < 24576) { src = eW2; base = 16384; }
    else if (i < 40960) { src = nW1; base = 24576; }
    else if (i < 49152) { src = nW2; base = 40960; }
    else if (i < 57344) { src = vW1; base = 49152; }
    else                { src = cW1; base = 57344; }
    int loc = i - base;
    int p = loc >> 7, n = loc & 127;
    pk[i] = pack2(src[(2 * p) * 128 + n], src[(2 * p + 1) * 128 + n]);
}

__global__ void egnn_count_kernel(const int* __restrict__ row,
                                  float* __restrict__ cnt, int E) {
    int i = blockIdx.x * blockDim.x + threadIdx.x;
    if (i < E) atomicAdd(&cnt[row[i]], 1.0f);
}

// h = h_in @ emb_W + emb_b  ([N,16] x [16,128])
__global__ void egnn_embed_kernel(const float* __restrict__ hin,
                                  const float* __restrict__ W,
                                  const float* __restrict__ b,
                                  float* __restrict__ hout, int N) {
    __shared__ float sh[16];
    int n = blockIdx.x;
    if (n >= N) return;
    if (threadIdx.x < 16) sh[threadIdx.x] = hin[n * 16 + threadIdx.x];
    __syncthreads();
    int j = threadIdx.x;
    float acc = b[j];
#pragma unroll
    for (int k = 0; k < 16; ++k) acc += sh[k] * W[k * 128 + j];
    hout[(size_t)n * 128 + j] = acc;
}

// ---------------------------------------------------------------------------
// Fused edge kernel: one wave per 16-edge tile.
//   feat = [h[row] | h[col]]   (bf16 staged in per-wave LDS, A layout)
//   z1   = relu(feat @ eW1[0:256] + eb1 + rel_dist*eW1[256])     (WMMA K=256)
//   m    = z1 @ eW2 + eb2                                        (WMMA K=128)
//   m_i[row] += m   (atomic scatter)
//   LAST layer only:
//   t    = relu(m @ cW1 + cb1);  coord = t . cW2 + cb2           (WMMA K=128)
//   scat[row] += rel_pos2 * coord
// ---------------------------------------------------------------------------
template <bool LAST>
__global__ __launch_bounds__(256) void egnn_edge_kernel(
    const float* __restrict__ hcur, const float* __restrict__ x,
    const int* __restrict__ eidx,
    const unsigned* __restrict__ pkE,   // [eW1pk | eW2pk] 24576 uints
    const unsigned* __restrict__ pkC,   // cW1pk 8192 uints
    const float* __restrict__ eW1f,     // f32 257x128 (row 256 used)
    const float* __restrict__ eb1, const float* __restrict__ eb2,
    const float* __restrict__ cb1, const float* __restrict__ cW2f,
    const float* __restrict__ cb2,
    float* __restrict__ m_i, float* __restrict__ scat, int E) {
    extern __shared__ unsigned smem[];   // 24576 (weights) + 8*2048 (staging)
    unsigned* sW = smem;
    load_weights_lds(sW, pkE);

    const int lane = threadIdx.x & 31;
    const int wv   = threadIdx.x >> 5;
    const int Nn   = lane & 15;          // == M for A operand, == N for B/C
    const int half = lane >> 4;
    unsigned* aB = smem + 24576 + wv * 2048;           // 16 edges x 128 pairs
    unsigned short* aB16 = (unsigned short*)aB;

    const int tiles = E >> 4;
    const int gw = blockIdx.x * 8 + wv;
    const int nw = gridDim.x * 8;
    const int* rowP = eidx;
    const int* colP = eidx + E;

    for (int t = gw; t < tiles; t += nw) {
        int ed = t * 16 + lane;
        int ec = ed < E ? ed : (E - 1);
        int valid = (lane < 16) && (ed < E);
        int rowI = rowP[ec];
        int colI = colP[ec];

        // prefetch next tile's edge indices (global_prefetch_b8)
        int t2 = t + nw;
        if (t2 < tiles && lane < 16) {
            __builtin_prefetch(&rowP[t2 * 16 + lane], 0, 0);
            __builtin_prefetch(&colP[t2 * 16 + lane], 0, 0);
        }

        float relD = 0.f, r2x = 0.f, r2y = 0.f;
        if (lane < 16) {
            float dx = x[rowI * 2 + 0] - x[colI * 2 + 0];
            float dy = x[rowI * 2 + 1] - x[colI * 2 + 1];
            relD = dx * dx + dy * dy;
            if (LAST) {
                float th = atan2f(dy, dx);
                r2x = relD * cosf(2.f * th);
                r2y = relD * sinf(2.f * th);
            }
        }

        // ---- stage features as bf16 pairs: [e][k/2], k<128 = h[row], k>=128 = h[col]
#pragma unroll
        for (int e = 0; e < 16; ++e) {
            int rn = __shfl(rowI, e);
            int cn = __shfl(colI, e);
            float4 a = ((const float4*)(hcur + (size_t)rn * 128))[lane];
            float4 b = ((const float4*)(hcur + (size_t)cn * 128))[lane];
            aB[e * 128 + lane * 2 + 0]      = pack2(a.x, a.y);
            aB[e * 128 + lane * 2 + 1]      = pack2(a.z, a.w);
            aB[e * 128 + 64 + lane * 2 + 0] = pack2(b.x, b.y);
            aB[e * 128 + 64 + lane * 2 + 1] = pack2(b.z, b.w);
        }

        // ---- GEMM1: K=256 in 8 k-tiles ----
        Frag16 af[8];
#pragma unroll
        for (int kt = 0; kt < 8; ++kt)
#pragma unroll
            for (int j = 0; j < 8; ++j) {
                int p = kt * 16 + half * 8 + j;
                af[kt].u[j] = aB[Nn * 128 + p];
            }
#pragma unroll
        for (int nt = 0; nt < 8; ++nt) {
            int n = nt * 16 + Nn;
            float b1v = eb1[n];
            float wl  = eW1f[256 * 128 + n];   // rel_dist column of eW1
            v8f acc;
#pragma unroll
            for (int r = 0; r < 8; ++r)
                acc[r] = b1v + __shfl(relD, r + half * 8) * wl;
#pragma unroll
            for (int kt = 0; kt < 8; ++kt) {
                Frag16 bf;
#pragma unroll
                for (int j = 0; j < 8; ++j)
                    bf.u[j] = sW[(kt * 16 + half * 8 + j) * 128 + n];
                acc = __builtin_amdgcn_wmma_f32_16x16x32_bf16(
                    false, af[kt].v, false, bf.v, (short)0, acc, false, false);
            }
#pragma unroll
            for (int r = 0; r < 8; ++r) {
                float z = acc[r] > 0.f ? acc[r] : 0.f;     // relu
                aB16[(r + half * 8) * 256 + n] = f2bf(z);  // restage z1
            }
        }

        // ---- GEMM2: m = z1 @ eW2 + eb2, K=128 ----
        Frag16 af2[4];
#pragma unroll
        for (int kt = 0; kt < 4; ++kt)
#pragma unroll
            for (int j = 0; j < 8; ++j) {
                int p = kt * 16 + half * 8 + j;
                af2[kt].u[j] = aB[Nn * 128 + p];
            }
        v8f m[8];
#pragma unroll
        for (int nt = 0; nt < 8; ++nt) {
            int n = nt * 16 + Nn;
            float b2v = eb2[n];
            v8f acc;
#pragma unroll
            for (int r = 0; r < 8; ++r) acc[r] = b2v;
#pragma unroll
            for (int kt = 0; kt < 4; ++kt) {
                Frag16 bf;
#pragma unroll
                for (int j = 0; j < 8; ++j)
                    bf.u[j] = sW[16384 + (kt * 16 + half * 8 + j) * 128 + n];
                acc = __builtin_amdgcn_wmma_f32_16x16x32_bf16(
                    false, af2[kt].v, false, bf.v, (short)0, acc, false, false);
            }
            m[nt] = acc;
        }

        // ---- scatter m into m_i[row]; stage m bf16 for coord MLP ----
#pragma unroll
        for (int r = 0; r < 8; ++r) {
            int M = r + half * 8;
            int nodeR = __shfl(rowI, M);
            int vM    = __shfl(valid, M);
#pragma unroll
            for (int nt = 0; nt < 8; ++nt) {
                float val = m[nt][r];
                if (vM) atomicAdd(&m_i[(size_t)nodeR * 128 + nt * 16 + Nn], val);
                if (LAST) aB16[M * 256 + nt * 16 + Nn] = f2bf(val);
            }
        }

        if (LAST) {
            // ---- GEMM3: t = relu(m @ cW1 + cb1); coord = t . cW2 + cb2 ----
            Frag16 af3[4];
#pragma unroll
            for (int kt = 0; kt < 4; ++kt)
#pragma unroll
                for (int j = 0; j < 8; ++j) {
                    int p = kt * 16 + half * 8 + j;
                    af3[kt].u[j] = aB[Nn * 128 + p];
                }
            float part[8];
#pragma unroll
            for (int r = 0; r < 8; ++r) part[r] = 0.f;
#pragma unroll
            for (int nt = 0; nt < 8; ++nt) {
                int n = nt * 16 + Nn;
                float b = cb1[n];
                v8f acc;
#pragma unroll
                for (int r = 0; r < 8; ++r) acc[r] = b;
#pragma unroll
                for (int kt = 0; kt < 4; ++kt) {
                    Frag16 bf;
#pragma unroll
                    for (int j = 0; j < 8; ++j)
                        bf.u[j] = pkC[(kt * 16 + half * 8 + j) * 128 + n];
                    acc = __builtin_amdgcn_wmma_f32_16x16x32_bf16(
                        false, af3[kt].v, false, bf.v, (short)0, acc, false, false);
                }
                float w2 = cW2f[n];
#pragma unroll
                for (int r = 0; r < 8; ++r) {
                    float tv = acc[r] > 0.f ? acc[r] : 0.f;
                    part[r] += tv * w2;
                }
            }
#pragma unroll
            for (int off = 8; off >= 1; off >>= 1)
#pragma unroll
                for (int r = 0; r < 8; ++r)
                    part[r] += __shfl_xor(part[r], off);
            float cb2v = cb2[0];
            float cE = 0.f;
#pragma unroll
            for (int e = 0; e < 16; ++e) {
                float ce = (e < 8) ? __shfl(part[e], 0) : __shfl(part[e - 8], 16);
                if (lane == e) cE = ce + cb2v;
            }
            if (valid) {
                atomicAdd(&scat[(size_t)rowI * 2 + 0], r2x * cE);
                atomicAdd(&scat[(size_t)rowI * 2 + 1], r2y * cE);
            }
        }
    }
}

// ---------------------------------------------------------------------------
// Fused node kernel: one wave per 16-node tile.
//   h += mlp([h | m_i])          (two WMMA chains, K=256 then K=128)
//   LAST layer: v = mlp(h); v += scat/denom; v = l2normalize(v)
// ---------------------------------------------------------------------------
template <bool LAST>
__global__ __launch_bounds__(256) void egnn_node_kernel(
    float* __restrict__ hcur, const float* __restrict__ m_i,
    const unsigned* __restrict__ pkN,   // [nW1pk | nW2pk] 24576 uints
    const unsigned* __restrict__ pkV,   // vW1pk 8192 uints
    const float* __restrict__ nb1, const float* __restrict__ nb2,
    const float* __restrict__ vb1, const float* __restrict__ vW2f,
    const float* __restrict__ vb2,
    const float* __restrict__ cnt, const float* __restrict__ scat,
    float* __restrict__ vout, int N) {
    extern __shared__ unsigned smem[];
    unsigned* sW = smem;
    load_weights_lds(sW, pkN);

    const int lane = threadIdx.x & 31;
    const int wv   = threadIdx.x >> 5;
    const int Nn   = lane & 15;
    const int half = lane >> 4;
    unsigned* aB = smem + 24576 + wv * 2048;
    unsigned short* aB16 = (unsigned short*)aB;

    const int tiles = (N + 15) >> 4;
    const int gw = blockIdx.x * 8 + wv;
    const int nw = gridDim.x * 8;

    for (int t = gw; t < tiles; t += nw) {
        int nd = t * 16 + lane;
        int nodeI = nd < N ? nd : (N - 1);
        int valid = (lane < 16) && (nd < N);

        // stage [h | m_i]
#pragma unroll
        for (int e = 0; e < 16; ++e) {
            int nn = __shfl(nodeI, e);
            float4 a = ((const float4*)(hcur + (size_t)nn * 128))[lane];
            float4 b = ((const float4*)(m_i  + (size_t)nn * 128))[lane];
            aB[e * 128 + lane * 2 + 0]      = pack2(a.x, a.y);
            aB[e * 128 + lane * 2 + 1]      = pack2(a.z, a.w);
            aB[e * 128 + 64 + lane * 2 + 0] = pack2(b.x, b.y);
            aB[e * 128 + 64 + lane * 2 + 1] = pack2(b.z, b.w);
        }

        // GEMM1: z = relu([h|m] @ nW1 + nb1), z staged into second half
        Frag16 af[8];
#pragma unroll
        for (int kt = 0; kt < 8; ++kt)
#pragma unroll
            for (int j = 0; j < 8; ++j) {
                int p = kt * 16 + half * 8 + j;
                af[kt].u[j] = aB[Nn * 128 + p];
            }
#pragma unroll
        for (int nt = 0; nt < 8; ++nt) {
            int n = nt * 16 + Nn;
            float b1v = nb1[n];
            v8f acc;
#pragma unroll
            for (int r = 0; r < 8; ++r) acc[r] = b1v;
#pragma unroll
            for (int kt = 0; kt < 8; ++kt) {
                Frag16 bf;
#pragma unroll
                for (int j = 0; j < 8; ++j)
                    bf.u[j] = sW[(kt * 16 + half * 8 + j) * 128 + n];
                acc = __builtin_amdgcn_wmma_f32_16x16x32_bf16(
                    false, af[kt].v, false, bf.v, (short)0, acc, false, false);
            }
#pragma unroll
            for (int r = 0; r < 8; ++r) {
                float z = acc[r] > 0.f ? acc[r] : 0.f;
                aB16[(r + half * 8) * 256 + 128 + n] = f2bf(z);
            }
        }

        // GEMM2: upd = z @ nW2 + nb2; h_new = h_old + upd
        Frag16 af2[4];
#pragma unroll
        for (int kt = 0; kt < 4; ++kt)
#pragma unroll
            for (int j = 0; j < 8; ++j) {
                int p = 64 + kt * 16 + half * 8 + j;
                af2[kt].u[j] = aB[Nn * 128 + p];
            }
#pragma unroll
        for (int nt = 0; nt < 8; ++nt) {
            int n = nt * 16 + Nn;
            float b2v = nb2[n];
            v8f acc;
#pragma unroll
            for (int r = 0; r < 8; ++r) acc[r] = b2v;
#pragma unroll
            for (int kt = 0; kt < 4; ++kt) {
                Frag16 bf;
#pragma unroll
                for (int j = 0; j < 8; ++j)
                    bf.u[j] = sW[16384 + (kt * 16 + half * 8 + j) * 128 + n];
                acc = __builtin_amdgcn_wmma_f32_16x16x32_bf16(
                    false, af2[kt].v, false, bf.v, (short)0, acc, false, false);
            }
#pragma unroll
            for (int r = 0; r < 8; ++r) {
                int M = r + half * 8;
                int nodeM = __shfl(nodeI, M);
                int vM    = __shfl(valid, M);
                float hn = hcur[(size_t)nodeM * 128 + n] + acc[r];
                if (vM) hcur[(size_t)nodeM * 128 + n] = hn;
                if (LAST) aB16[M * 256 + n] = f2bf(hn);   // stage h_new, first half
            }
        }

        if (LAST) {
            // v = mlp(h_new): t = relu(h_new @ vW1 + vb1); v = t @ vW2 + vb2
            Frag16 af3[4];
#pragma unroll
            for (int kt = 0; kt < 4; ++kt)
#pragma unroll
                for (int j = 0; j < 8; ++j) {
                    int p = kt * 16 + half * 8 + j;
                    af3[kt].u[j] = aB[Nn * 128 + p];
                }
            float p0[8], p1[8];
#pragma unroll
            for (int r = 0; r < 8; ++r) { p0[r] = 0.f; p1[r] = 0.f; }
#pragma unroll
            for (int nt = 0; nt < 8; ++nt) {
                int n = nt * 16 + Nn;
                float b = vb1[n];
                v8f acc;
#pragma unroll
                for (int r = 0; r < 8; ++r) acc[r] = b;
#pragma unroll
                for (int kt = 0; kt < 4; ++kt) {
                    Frag16 bf;
#pragma unroll
                    for (int j = 0; j < 8; ++j)
                        bf.u[j] = pkV[(kt * 16 + half * 8 + j) * 128 + n];
                    acc = __builtin_amdgcn_wmma_f32_16x16x32_bf16(
                        false, af3[kt].v, false, bf.v, (short)0, acc, false, false);
                }
                float w0 = vW2f[n * 2 + 0], w1 = vW2f[n * 2 + 1];
#pragma unroll
                for (int r = 0; r < 8; ++r) {
                    float tv = acc[r] > 0.f ? acc[r] : 0.f;
                    p0[r] += tv * w0;
                    p1[r] += tv * w1;
                }
            }
#pragma unroll
            for (int off = 8; off >= 1; off >>= 1)
#pragma unroll
                for (int r = 0; r < 8; ++r) {
                    p0[r] += __shfl_xor(p0[r], off);
                    p1[r] += __shfl_xor(p1[r], off);
                }
            float vb20 = vb2[0], vb21 = vb2[1];
            float v0 = 0.f, v1 = 0.f;
#pragma unroll
            for (int e = 0; e < 16; ++e) {
                float a0 = (e < 8) ? __shfl(p0[e], 0) : __shfl(p0[e - 8], 16);
                float a1 = (e < 8) ? __shfl(p1[e], 0) : __shfl(p1[e - 8], 16);
                if (lane == e) { v0 = a0 + vb20; v1 = a1 + vb21; }
            }
            if (valid) {
                float c = cnt[nodeI];
                float den = c > 1.f ? c : 1.f;
                v0 += scat[(size_t)nodeI * 2 + 0] / den;
                v1 += scat[(size_t)nodeI * 2 + 1] / den;
                float nrm = sqrtf(v0 * v0 + v1 * v1);
                float inv = 1.f / fmaxf(nrm, 1e-12f);
                vout[(size_t)nodeI * 2 + 0] = v0 * inv;
                vout[(size_t)nodeI * 2 + 1] = v1 * inv;
            }
        }
    }
}

// ---------------------------------------------------------------------------
static inline size_t align256(size_t v) { return (v + 255) & ~(size_t)255; }

extern "C" void kernel_launch(void* const* d_in, const int* in_sizes, int n_in,
                              void* d_out, int out_size, void* d_ws, size_t ws_size,
                              hipStream_t stream) {
    (void)n_in; (void)out_size; (void)ws_size;
    const float* h_in  = (const float*)d_in[0];
    const float* x     = (const float*)d_in[1];
    const int*   eidx  = (const int*)d_in[2];
    const float* emb_W = (const float*)d_in[3];
    const float* emb_b = (const float*)d_in[4];
    const float* e_W1  = (const float*)d_in[5];
    const float* e_b1  = (const float*)d_in[6];
    const float* e_W2  = (const float*)d_in[7];
    const float* e_b2  = (const float*)d_in[8];
    const float* n_W1  = (const float*)d_in[9];
    const float* n_b1  = (const float*)d_in[10];
    const float* n_W2  = (const float*)d_in[11];
    const float* n_b2  = (const float*)d_in[12];
    const float* v_W1  = (const float*)d_in[13];
    const float* v_b1  = (const float*)d_in[14];
    const float* v_W2  = (const float*)d_in[15];
    const float* v_b2  = (const float*)d_in[16];
    const float* c_W1  = (const float*)d_in[17];
    const float* c_b1  = (const float*)d_in[18];
    const float* c_W2  = (const float*)d_in[19];
    const float* c_b2  = (const float*)d_in[20];

    const int N = in_sizes[0] / 16;   // P = 16
    const int E = in_sizes[2] / 2;

    float* hout = (float*)d_out;                  // [N,128]
    float* xout = hout + (size_t)N * 128;         // [N,2]
    float* vout = xout + (size_t)N * 2;           // [N,2]

    // Workspace layout
    unsigned* pk = (unsigned*)d_ws;               // 65536 uints packed bf16 weights
    size_t off = align256((size_t)65536 * 4);
    float* cnt  = (float*)((char*)d_ws + off); off += align256((size_t)N * 4);
    float* scat = (float*)((char*)d_ws + off); off += align256((size_t)N * 8);
    float* m_i  = (float*)((char*)d_ws + off);    // [N,128]

    const unsigned* pkE = pk;          // eW1 + eW2
    const unsigned* pkN = pk + 24576;  // nW1 + nW2
    const unsigned* pkV = pk + 49152;  // vW1
    const unsigned* pkC = pk + 57344;  // cW1

    const size_t ldsBytes = 163840;    // 96KB weights + 8 waves x 8KB staging

    hipMemsetAsync(cnt, 0, (size_t)N * 4, stream);
    hipMemsetAsync(scat, 0, (size_t)N * 8, stream);

    egnn_pack_kernel<<<256, 256, 0, stream>>>(e_W1, e_W2, n_W1, n_W2, v_W1, c_W1, pk);
    egnn_count_kernel<<<(E + 255) / 256, 256, 0, stream>>>(eidx, cnt, E);
    egnn_embed_kernel<<<N, 128, 0, stream>>>(h_in, emb_W, emb_b, hout, N);

    const int edgeBlocks = 2048;
    const int nodeTiles  = (N + 15) / 16;
    const int nodeBlocks = (nodeTiles + 7) / 8;

    // ---- layer 0 (coord / v MLPs are dead: overwritten by layer 1) ----
    hipMemsetAsync(m_i, 0, (size_t)N * 512, stream);
    egnn_edge_kernel<false><<<edgeBlocks, 256, ldsBytes, stream>>>(
        hout, x, eidx, pkE, pkC, e_W1, e_b1, e_b2, c_b1, c_W2, c_b2, m_i, scat, E);
    egnn_node_kernel<false><<<nodeBlocks, 256, ldsBytes, stream>>>(
        hout, m_i, pkN, pkV, n_b1, n_b2, v_b1, v_W2, v_b2, cnt, scat, vout, N);

    // ---- layer 1 (last): full pipeline incl. coord scatter + v output ----
    hipMemsetAsync(m_i, 0, (size_t)N * 512, stream);
    egnn_edge_kernel<true><<<edgeBlocks, 256, ldsBytes, stream>>>(
        hout, x, eidx, pkE, pkC, e_W1, e_b1, e_b2, c_b1, c_W2, c_b2, m_i, scat, E);
    egnn_node_kernel<true><<<nodeBlocks, 256, ldsBytes, stream>>>(
        hout, m_i, pkN, pkV, n_b1, n_b2, v_b1, v_W2, v_b2, cnt, scat, vout, N);

    // x passes through unchanged
    hipMemcpyAsync(xout, x, (size_t)N * 8, hipMemcpyDeviceToDevice, stream);
}
